// HybridDecoder_80582176407876
// MI455X (gfx1250) — matrix operand order repaired and checked by reading, not compiled, not run on hardware
//
#include <hip/hip_runtime.h>

#define B_ 64
#define T_ 32
#define H_ 512
#define S_ 400

typedef __attribute__((ext_vector_type(16))) __bf16 v16bf;
typedef __attribute__((ext_vector_type(8)))  float  v8f;
typedef __attribute__((ext_vector_type(4)))  unsigned int v4u;
typedef __attribute__((ext_vector_type(8)))  int    v8i;
typedef __attribute__((ext_vector_type(4)))  int    v4i;
typedef unsigned short ushort_t;
typedef unsigned int   uint_t;

// ---------------- helpers ----------------
__device__ __forceinline__ ushort_t f2bf(float f) {
  uint_t u = __float_as_uint(f);
  u += 0x7fffu + ((u >> 16) & 1u);          // round-to-nearest-even
  return (ushort_t)(u >> 16);
}
__device__ __forceinline__ float sigm(float x) {
  return 1.0f / (1.0f + __expf(-x));
}

// Issue one TDM 2D tile load: tile[64 rows x 32 cols] of bf16 from a row-major
// [rows][rowStride] tensor into contiguous LDS. D# per CDNA5 ISA.
__device__ __forceinline__ void tdm_load_tile(unsigned ldsOff,
                                              const ushort_t* gaddr,
                                              unsigned remD0, unsigned remD1,
                                              unsigned stride0) {
  unsigned long long ga = (unsigned long long)(uintptr_t)gaddr;
  v4u g0;
  g0.x = 1u;                                        // count=1 (valid user D#)
  g0.y = ldsOff;                                    // lds_addr (bytes)
  g0.z = (uint_t)(ga & 0xffffffffu);                // global_addr[31:0]
  g0.w = (uint_t)((ga >> 32) & 0x1ffffffu) | (2u << 30);  // addr[56:32], type=2
  v8i g1;
  g1[0] = (int)(1u << 16);                          // data_size=1 (2 bytes)
  g1[1] = (int)((remD0 & 0xffffu) << 16);           // tensor_dim0[15:0]
  g1[2] = (int)((remD0 >> 16) | ((remD1 & 0xffffu) << 16));   // d0 hi | d1 lo
  g1[3] = (int)((remD1 >> 16) | (32u << 16));       // d1 hi | tile_dim0=32
  g1[4] = (int)64u;                                 // tile_dim1=64, tile_dim2=0
  g1[5] = (int)stride0;                             // tensor_dim0_stride[31:0]
  g1[6] = 0;
  g1[7] = 0;
  v4i gz4 = {0, 0, 0, 0};                           // 2D tensor: groups 2/3 unused
  v8i gz8 = {0, 0, 0, 0, 0, 0, 0, 0};
  __builtin_amdgcn_tensor_load_to_lds(g0, g1, gz4, gz4, gz8, 0);
}

// ---------------- workspace layout (bytes) ----------------
static constexpr size_t OFF_WG   = 0;                                  // WgT [2048][1536] bf16 ([Wx;Wh]^T)
static constexpr size_t OFF_WIN  = OFF_WG   + (size_t)2048*1536*2;     // WinT [512][512] bf16
static constexpr size_t OFF_WOUT = OFF_WIN  + (size_t)512*512*2;       // WoutT [512][2048] bf16
static constexpr size_t OFF_ABF  = OFF_WOUT + (size_t)512*2048*2;      // [2][64][1536] bf16
static constexpr size_t OFF_HBF  = OFF_ABF  + (size_t)2*64*1536*2;     // [2][64][512]  bf16
static constexpr size_t OFF_COMB = OFF_HBF  + (size_t)2*64*512*2;      // [64][2048]    bf16
static constexpr size_t OFF_G    = OFF_COMB + (size_t)64*2048*2;       // [2][64][2048] f32
static constexpr size_t OFF_TGT  = OFF_G    + (size_t)2*64*2048*4;     // [2][64][512]  f32
static constexpr size_t OFF_TH   = OFF_TGT  + (size_t)2*64*512*4;      // [64][512] f32 x5
static constexpr size_t OFF_TC   = OFF_TH   + (size_t)64*512*4;
static constexpr size_t OFF_XH   = OFF_TC   + (size_t)64*512*4;
static constexpr size_t OFF_XC   = OFF_XH   + (size_t)64*512*4;
static constexpr size_t OFF_OUTS = OFF_XC   + (size_t)64*512*4;

// ---------------- prologue kernels ----------------
// transpose + convert: src f32 [K][N] row-major -> dst bf16 [N][dstStride] at col dstOff
__global__ void cvt_T_bf16(const float* __restrict__ src, ushort_t* __restrict__ dst,
                           int K, int N, int dstStride, int dstOff) {
  int idx = blockIdx.x * 256 + threadIdx.x;
  if (idx >= K * N) return;
  int k = idx / N, n = idx % N;                    // coalesced reads
  dst[(size_t)n * dstStride + dstOff + k] = f2bf(src[idx]);
}
__global__ void copy_f32(const float* __restrict__ src,
                         float* __restrict__ dst, int n) {
  int i = blockIdx.x * 256 + threadIdx.x;
  if (i < n) dst[i] = src[i];
}

// ---------------- per-step: build A = [emb | out | h_branch] in bf16 ----------------
__global__ void pack_A_kernel(int t, const int* __restrict__ inputs,
                              const float* __restrict__ emb0,
                              const float* __restrict__ word_emb,
                              const float* __restrict__ outS,
                              const float* __restrict__ thS,
                              const float* __restrict__ xhS,
                              ushort_t* __restrict__ Abf) {
  int idx = blockIdx.x * 256 + threadIdx.x;            // < 2*64*1536
  int br  = idx / (B_ * 3 * H_);
  int rem = idx % (B_ * 3 * H_);
  int b   = rem / (3 * H_);
  int k   = rem % (3 * H_);
  float v;
  if (k < H_) {
    // scan semantics: step t consumes the embedding carried from step t-1
    v = (t == 0) ? emb0[b * H_ + k]
                 : word_emb[(size_t)inputs[b * T_ + (t - 1)] * H_ + k];
  } else if (k < 2 * H_) {
    v = outS[b * H_ + (k - H_)];
  } else {
    v = (br ? xhS : thS)[b * H_ + (k - 2 * H_)];
  }
  Abf[(size_t)idx] = f2bf(v);
}

// ---------------- WMMA GEMM: C[64,N] f32 = A[64,K] bf16 x B^T[N,K] bf16 ----------------
// block = 256 threads (8 waves), computes full M=64 x Ntile=64.
// Slabs are DMA'd by the Tensor Data Mover into double-buffered LDS.
// mode 0: plain store. mode 1: tanh, store to C and strided out2 (d_out slice).
__global__ __launch_bounds__(256)
void wmma_gemm_kernel(const ushort_t* __restrict__ A,
                      const ushort_t* __restrict__ Bt,
                      float* __restrict__ C,
                      int K, int N, long long aBatch, long long cBatch,
                      int mode, float* __restrict__ out2, int out2RowStride) {
  __shared__ ushort_t As[2][64 * 32];   // [m][k] row-major slab, double buffered
  __shared__ ushort_t Bs[2][64 * 32];   // [n][k] slab (B already transposed in HBM)
  const int tid   = threadIdx.x;
  const int lane  = tid & 31;
  const int w     = tid >> 5;
  const int n0    = blockIdx.x * 64;
  const int batch = blockIdx.y;
  const ushort_t* Ab = A + (size_t)batch * aBatch;
  const ushort_t* Bb = Bt + (size_t)n0 * K;
  const unsigned asOff = (unsigned)(uintptr_t)&As[0][0];
  const unsigned bsOff = (unsigned)(uintptr_t)&Bs[0][0];

  const int mt  = w & 3;             // M tile 0..3
  const int ntA = (w >> 2) * 2;      // N tile base (two tiles per wave)
  v8f acc0 = {}; v8f acc1 = {};

  const int mrow   = mt * 16 + (lane & 15);
  const int kbaseA = (lane >> 4) << 4;   // 0 / 16  (A K-halves per lane group)
  const int kbaseB = (lane >> 4) << 4;   // 0 / 16  (B K-halves per lane group)
  const int col0   = ntA * 16 + (lane & 15);
  const int col1   = col0 + 16;

  const int nk = K >> 5;
  // prime buffer 0
  if (w == 0) {
    tdm_load_tile(asOff, Ab, (unsigned)K, 64u, (unsigned)K);
    tdm_load_tile(bsOff, Bb, (unsigned)K, (unsigned)(N - n0), (unsigned)K);
  }
  for (int kt = 0; kt < nk; ++kt) {
    const int buf = kt & 1;
    if (w == 0) {
      if (kt + 1 < nk) {                 // stage next slab into other buffer
        const int k1 = (kt + 1) << 5;
        tdm_load_tile(asOff + (unsigned)((buf ^ 1) * 4096), Ab + k1,
                      (unsigned)(K - k1), 64u, (unsigned)K);
        tdm_load_tile(bsOff + (unsigned)((buf ^ 1) * 4096), Bb + k1,
                      (unsigned)(K - k1), (unsigned)(N - n0), (unsigned)K);
        __builtin_amdgcn_s_wait_tensorcnt(2);   // current buffer complete
      } else {
        __builtin_amdgcn_s_wait_tensorcnt(0);
      }
    }
    __syncthreads();

    const uint_t* Asb = (const uint_t*)As[buf];
    const uint_t* Bsb = (const uint_t*)Bs[buf];
    union { v16bf v; uint_t u[8]; } af, b0, b1;
#pragma unroll
    for (int j = 0; j < 8; ++j) {
      int kk = kbaseA + (j << 1);
      af.u[j] = Asb[mrow * 16 + (kk >> 1)];
    }
#pragma unroll
    for (int j = 0; j < 8; ++j) {
      b0.u[j] = Bsb[col0 * 16 + (kbaseB >> 1) + j];
      b1.u[j] = Bsb[col1 * 16 + (kbaseB >> 1) + j];
    }
    acc0 = __builtin_amdgcn_wmma_f32_16x16x32_bf16(false, af.v, false, b0.v,
                                                   (short)0, acc0, false, false);
    acc1 = __builtin_amdgcn_wmma_f32_16x16x32_bf16(false, af.v, false, b1.v,
                                                   (short)0, acc1, false, false);
    __syncthreads();   // all waves done reading before TDM rewrites this buffer
  }

  // D layout: VGPR r -> row = r + (lane>=16 ? 8 : 0), col = lane&15
  float* Cb = C + (size_t)batch * cBatch;
  const int rbase = mt * 16 + ((lane >> 4) << 3);
  const int gcol0 = n0 + col0;
#pragma unroll
  for (int r = 0; r < 8; ++r) {
    int grow = rbase + r;
    if (mode == 0) {
      Cb[(size_t)grow * N + gcol0]      = acc0[r];
      Cb[(size_t)grow * N + gcol0 + 16] = acc1[r];
    } else {
      float v0 = tanhf(acc0[r]);
      float v1 = tanhf(acc1[r]);
      Cb[(size_t)grow * N + gcol0]      = v0;
      Cb[(size_t)grow * N + gcol0 + 16] = v1;
      out2[(size_t)grow * out2RowStride + gcol0]      = v0;
      out2[(size_t)grow * out2RowStride + gcol0 + 16] = v1;
    }
  }
}

// ---------------- LSTM pointwise (both branches) ----------------
__global__ void lstm_kernel(const float* __restrict__ g,
                            const float* __restrict__ bias,
                            float* __restrict__ thS, float* __restrict__ tcS,
                            float* __restrict__ xhS, float* __restrict__ xcS,
                            ushort_t* __restrict__ hbf,
                            ushort_t* __restrict__ comb) {
  int idx = blockIdx.x * 256 + threadIdx.x;            // < 2*64*512
  int br  = idx / (B_ * H_);
  int rem = idx % (B_ * H_);
  int b   = rem / H_;
  int h   = rem % H_;
  const float* gr = g + ((size_t)br * B_ + b) * (4 * H_);
  float gi = gr[h]           + bias[h];
  float gf = gr[H_ + h]      + bias[H_ + h];
  float gg = gr[2 * H_ + h]  + bias[2 * H_ + h];
  float go = gr[3 * H_ + h]  + bias[3 * H_ + h];
  float* cS = br ? xcS : tcS;
  float* hS = br ? xhS : thS;
  float c_new = sigm(gf) * cS[rem] + sigm(gi) * tanhf(gg);
  float h_new = sigm(go) * tanhf(c_new);
  cS[rem] = c_new;
  hS[rem] = h_new;
  ushort_t hb = f2bf(h_new);
  hbf[idx] = hb;
  // combined = [w_tree, w_text, th, xh]
  comb[(size_t)b * (4 * H_) + 2 * H_ + br * H_ + h] = hb;
}

// ---------------- attention: scores, softmax, weighted ctx ----------------
__global__ __launch_bounds__(256)
void attend_kernel(const float* __restrict__ tgt,
                   const float* __restrict__ tree_ctx,
                   const float* __restrict__ text_ctx,
                   ushort_t* __restrict__ comb) {
  __shared__ float tg[H_];
  __shared__ float sc[S_];
  __shared__ float redA[8];
  __shared__ float redB[8];
  const int tid  = threadIdx.x;
  const int lane = tid & 31;
  const int w    = tid >> 5;
  const int b    = blockIdx.x;
  const int br   = blockIdx.y;
  const float* ctx = (br ? text_ctx : tree_ctx) + (size_t)b * S_ * H_;
  const float* tgb = tgt + ((size_t)br * B_ + b) * H_;
  tg[tid]       = tgb[tid];
  tg[tid + 256] = tgb[tid + 256];
  __syncthreads();

  // scores: one row per wave, lane-parallel dot + wave32 shuffle reduce
  for (int s = w; s < S_; s += 8) {
    const float* row = ctx + (size_t)s * H_;
    float sum = 0.f;
#pragma unroll 4
    for (int h = lane; h < H_; h += 32) sum += row[h] * tg[h];
#pragma unroll
    for (int off = 16; off; off >>= 1) sum += __shfl_xor(sum, off, 32);
    if (lane == 0) sc[s] = sum;
  }
  __syncthreads();

  // softmax: block max
  float m = -3.4e38f;
  for (int s = tid; s < S_; s += 256) m = fmaxf(m, sc[s]);
#pragma unroll
  for (int off = 16; off; off >>= 1) m = fmaxf(m, __shfl_xor(m, off, 32));
  if (lane == 0) redA[w] = m;
  __syncthreads();
  m = redA[0];
#pragma unroll
  for (int i = 1; i < 8; ++i) m = fmaxf(m, redA[i]);

  // exp + block sum (each thread touches only its own strided subset of sc)
  float ls = 0.f;
  for (int s = tid; s < S_; s += 256) { float e = __expf(sc[s] - m); sc[s] = e; ls += e; }
#pragma unroll
  for (int off = 16; off; off >>= 1) ls += __shfl_xor(ls, off, 32);
  if (lane == 0) redB[w] = ls;
  __syncthreads();
  float tot = 0.f;
#pragma unroll
  for (int i = 0; i < 8; ++i) tot += redB[i];
  const float inv = 1.0f / tot;

  // weighted context: threads parallel over h (coalesced per s)
#pragma unroll
  for (int hh = 0; hh < 2; ++hh) {
    int h = tid + hh * 256;
    float acc = 0.f;
    for (int s = 0; s < S_; ++s) acc += sc[s] * ctx[(size_t)s * H_ + h];
    comb[(size_t)b * (4 * H_) + br * H_ + h] = f2bf(acc * inv);
  }
}

// ---------------- host ----------------
extern "C" void kernel_launch(void* const* d_in, const int* in_sizes, int n_in,
                              void* d_out, int out_size, void* d_ws, size_t ws_size,
                              hipStream_t stream) {
  const int*   inputs   = (const int*)  d_in[0];
  const float* emb0     = (const float*)d_in[1];
  const float* output0  = (const float*)d_in[2];
  const float* tree_h0  = (const float*)d_in[3];
  const float* tree_c0  = (const float*)d_in[4];
  const float* tree_ctx = (const float*)d_in[5];
  const float* text_h0  = (const float*)d_in[6];
  const float* text_c0  = (const float*)d_in[7];
  const float* text_ctx = (const float*)d_in[8];
  const float* word_emb = (const float*)d_in[9];
  const float* Wx       = (const float*)d_in[10];
  const float* Wh       = (const float*)d_in[11];
  const float* bias     = (const float*)d_in[12];
  const float* W_in     = (const float*)d_in[13];
  const float* W_out    = (const float*)d_in[14];
  float* out = (float*)d_out;

  char* ws = (char*)d_ws;
  ushort_t* WgT   = (ushort_t*)(ws + OFF_WG);    // [2048][1536]
  ushort_t* WinT  = (ushort_t*)(ws + OFF_WIN);   // [512][512]
  ushort_t* WoutT = (ushort_t*)(ws + OFF_WOUT);  // [512][2048]
  ushort_t* Abf   = (ushort_t*)(ws + OFF_ABF);
  ushort_t* hbf   = (ushort_t*)(ws + OFF_HBF);
  ushort_t* comb  = (ushort_t*)(ws + OFF_COMB);
  float* gbuf = (float*)(ws + OFF_G);
  float* tgt  = (float*)(ws + OFF_TGT);
  float* thS  = (float*)(ws + OFF_TH);
  float* tcS  = (float*)(ws + OFF_TC);
  float* xhS  = (float*)(ws + OFF_XH);
  float* xcS  = (float*)(ws + OFF_XC);
  float* outS = (float*)(ws + OFF_OUTS);

  // prologue: transposed bf16 weight packs ([Wx;Wh] stacked then ^T) + state init
  cvt_T_bf16<<<(1024 * 2048 + 255) / 256, 256, 0, stream>>>(Wx, WgT, 1024, 2048, 1536, 0);
  cvt_T_bf16<<<(512 * 2048 + 255) / 256, 256, 0, stream>>>(Wh, WgT, 512, 2048, 1536, 1024);
  cvt_T_bf16<<<(512 * 512 + 255) / 256, 256, 0, stream>>>(W_in, WinT, 512, 512, 512, 0);
  cvt_T_bf16<<<(2048 * 512 + 255) / 256, 256, 0, stream>>>(W_out, WoutT, 2048, 512, 2048, 0);
  const int NS = B_ * H_;
  copy_f32<<<NS / 256, 256, 0, stream>>>(tree_h0, thS, NS);
  copy_f32<<<NS / 256, 256, 0, stream>>>(tree_c0, tcS, NS);
  copy_f32<<<NS / 256, 256, 0, stream>>>(text_h0, xhS, NS);
  copy_f32<<<NS / 256, 256, 0, stream>>>(text_c0, xcS, NS);
  copy_f32<<<NS / 256, 256, 0, stream>>>(output0, outS, NS);

  for (int t = 0; t < T_; ++t) {
    pack_A_kernel<<<(2 * B_ * 3 * H_) / 256, 256, 0, stream>>>(
        t, inputs, emb0, word_emb, outS, thS, xhS, Abf);
    // gates: [64,1536] x [1536,2048] per branch (B given as B^T [2048][1536])
    wmma_gemm_kernel<<<dim3(2048 / 64, 2), 256, 0, stream>>>(
        Abf, WgT, gbuf, 1536, 2048, (long long)64 * 1536, (long long)64 * 2048,
        0, nullptr, 0);
    lstm_kernel<<<(2 * B_ * H_) / 256, 256, 0, stream>>>(
        gbuf, bias, thS, tcS, xhS, xcS, hbf, comb);
    // target: h @ W_in per branch
    wmma_gemm_kernel<<<dim3(512 / 64, 2), 256, 0, stream>>>(
        hbf, WinT, tgt, 512, 512, (long long)64 * 512, (long long)64 * 512,
        0, nullptr, 0);
    attend_kernel<<<dim3(B_, 2), 256, 0, stream>>>(tgt, tree_ctx, text_ctx, comb);
    // out: combined @ W_out, tanh, -> outS state + d_out[:, t, :]
    wmma_gemm_kernel<<<dim3(512 / 64, 1), 256, 0, stream>>>(
        comb, WoutT, outS, 2048, 512, 0, 0,
        1, out + (size_t)t * H_, T_ * H_);
  }
}
